// VisionPerformerTIMM_25529285607500
// MI455X (gfx1250) — matrix operand order repaired and checked by reading, not compiled, not run on hardware
//
#include <hip/hip_runtime.h>
#include <hip/hip_bf16.h>

// ============================================================================
// VisionPerformer (ViT-B + Performer attention) forward for MI455X (gfx1250).
// Big GEMMs run on v_wmma_f32_16x16x32_f16 with double-buffered LDS fed by
// GLOBAL_LOAD_ASYNC_TO_LDS_B128 (ASYNCcnt) so copies overlap matrix math.
// ============================================================================

typedef __attribute__((ext_vector_type(16))) _Float16 v16h;
typedef __attribute__((ext_vector_type(8)))  _Float16 v8h;
typedef __attribute__((ext_vector_type(8)))  float    v8f;

constexpr int L_    = 12;
constexpr int C_    = 768;
constexpr int H_    = 12;
constexpr int DH_   = 64;
constexpr int M_    = 64;
constexpr int NP_   = 197;
constexpr int HID_  = 3072;
constexpr int NCLS_ = 1000;
constexpr int B_    = 32;
constexpr int ROWS_  = B_ * NP_;   // 6304 token rows
constexpr int PROWS_ = B_ * 196;   // 6272 patch rows
constexpr int QKVN_  = 3 * C_;     // 2304

// ---------------- helpers ----------------
static __device__ __forceinline__ unsigned f2ord(float f) {
  unsigned u = __float_as_uint(f);
  return (u & 0x80000000u) ? ~u : (u | 0x80000000u);
}
static __device__ __forceinline__ float ord2f(unsigned k) {
  unsigned u = (k & 0x80000000u) ? (k ^ 0x80000000u) : ~k;
  return __uint_as_float(u);
}

// Async copy: 16 bytes global -> LDS per lane, tracked by ASYNCcnt.
static __device__ __forceinline__ void async_b128(unsigned lds_off, const _Float16* g) {
  asm volatile("global_load_async_to_lds_b128 %0, %1, off"
               :: "v"(lds_off), "v"(g) : "memory");
}
#define S_WAIT_ASYNC(n) asm volatile("s_wait_asynccnt %0" :: "i"(n) : "memory")

static __device__ __forceinline__ unsigned lds_off_u32(const void* p) {
  // LDS aperture lives in addr[63:32]; low 32 bits are the wave-relative offset.
  return (unsigned)(unsigned long long)p;
}

// ---------------- f32 -> f16 convert (weights, per layer) ----------------
__global__ void k_cvt_f16(const float* __restrict__ W, _Float16* __restrict__ W16, int n) {
  int i = blockIdx.x * blockDim.x + threadIdx.x;
  if (i < n) W16[i] = (_Float16)W[i];
}

// ============================================================================
// WMMA GEMM:  C[rows][N] = epilogue( A[rows][K] * W[N][K]^T + bias[N] )
//   EPI=0: C (f32) = acc + bias
//   EPI=1: Ch (f16) = gelu(acc + bias)          (exact GELU, erf)
//   EPI=2: C (f32) += acc + bias                (residual accumulate)
// Workgroup = 256 threads = 8 waves; tile 128(M) x 128(N); K step 32.
// Wave grid 4x2; each wave owns 32x64 = 2x4 WMMA tiles (8 wmma / K-step).
// LDS is double buffered and filled with global_load_async_to_lds_b128;
// out-of-range rows are address-clamped (their products are never stored).
// K must be a multiple of 32 (true for all call sites: 768, 3072).
// ============================================================================
template <int EPI>
__global__ __launch_bounds__(256)
void k_gemm(const _Float16* __restrict__ A,   // [rows][K]
            const _Float16* __restrict__ W,   // [N][K]
            const float*    __restrict__ bias,// [N]
            float*          __restrict__ Cf,  // EPI 0/2
            _Float16*       __restrict__ Ch,  // EPI 1
            int rows, int N, int K) {
  constexpr int PADK = 40;                      // 32 + 8-half pad (bank spread)
  __shared__ alignas(16) _Float16 As[2][128][PADK];
  __shared__ alignas(16) _Float16 Ws[2][128][PADK];

  const int tid  = threadIdx.x;
  const int lane = tid & 31;
  const int wave = tid >> 5;
  const int mBase = blockIdx.y * 128;
  const int nBase = blockIdx.x * 128;
  const int wm = (wave >> 1) * 32;   // 0,32,64,96
  const int wn = (wave & 1) * 64;    // 0,64
  const int l16 = lane & 15;
  const int g   = lane >> 4;

  v8f acc[2][4] = {};

  // Each thread async-copies 2 x b128 of A and 2 x b128 of W per K-tile.
  // slot s -> row = s>>2, 8-half chunk = s&3;  slots {tid, tid+256}.
  const int r0 = tid >> 2,         c0 = (tid & 3) * 8;
  const int r1 = (tid + 256) >> 2, c1 = ((tid + 256) & 3) * 8;
  const _Float16* Ap0 = A + (size_t)min(mBase + r0, rows - 1) * K + c0;
  const _Float16* Ap1 = A + (size_t)min(mBase + r1, rows - 1) * K + c1;
  const _Float16* Wp0 = W + (size_t)min(nBase + r0, N - 1) * K + c0;
  const _Float16* Wp1 = W + (size_t)min(nBase + r1, N - 1) * K + c1;
  unsigned lA0[2], lA1[2], lW0[2], lW1[2];
#pragma unroll
  for (int d = 0; d < 2; ++d) {
    lA0[d] = lds_off_u32(&As[d][r0][c0]);
    lA1[d] = lds_off_u32(&As[d][r1][c1]);
    lW0[d] = lds_off_u32(&Ws[d][r0][c0]);
    lW1[d] = lds_off_u32(&Ws[d][r1][c1]);
  }
  auto issue = [&](int buf, int k0) {
    async_b128(lA0[buf], Ap0 + k0);
    async_b128(lA1[buf], Ap1 + k0);
    async_b128(lW0[buf], Wp0 + k0);
    async_b128(lW1[buf], Wp1 + k0);
  };

  const int nk = K / 32;
  issue(0, 0);
  for (int kt = 0; kt < nk; ++kt) {
    const int cur = kt & 1;
    if (kt + 1 < nk) {
      issue(cur ^ 1, (kt + 1) * 32);  // prefetch next tile into other buffer
      S_WAIT_ASYNC(4);                // current tile's 4 copies complete
    } else {
      S_WAIT_ASYNC(0);
    }
    __syncthreads();

    // --- fragments per ISA layout (wave32) ---
    // A: lane=row, halves 0..7 -> K=8g..8g+7, halves 8..15 -> K=16+8g..
    // B: lane=n (lane%16), same K order as A: halves 0..7 -> K=8g.., 8..15 -> K=16+8g..
    union U { v16h v; v8h h[2]; };
    U af[2], bf[4];
#pragma unroll
    for (int mi = 0; mi < 2; ++mi) {
      af[mi].h[0] = *(const v8h*)&As[cur][wm + mi * 16 + l16][8 * g];
      af[mi].h[1] = *(const v8h*)&As[cur][wm + mi * 16 + l16][16 + 8 * g];
    }
#pragma unroll
    for (int ni = 0; ni < 4; ++ni) {
      bf[ni].h[0] = *(const v8h*)&Ws[cur][wn + ni * 16 + l16][8 * g];
      bf[ni].h[1] = *(const v8h*)&Ws[cur][wn + ni * 16 + l16][16 + 8 * g];
    }
#pragma unroll
    for (int mi = 0; mi < 2; ++mi)
#pragma unroll
      for (int ni = 0; ni < 4; ++ni)
        acc[mi][ni] = __builtin_amdgcn_wmma_f32_16x16x32_f16(
            false, af[mi].v, false, bf[ni].v, (short)0, acc[mi][ni], false, false);
    __syncthreads();                  // all waves done reading buf[cur]
  }

  // --- epilogue: D layout m = vgpr + 8*(lane/16), n = lane%16 ---
#pragma unroll
  for (int mi = 0; mi < 2; ++mi)
#pragma unroll
    for (int ni = 0; ni < 4; ++ni) {
      int n = nBase + wn + ni * 16 + l16;
      if (n >= N) continue;
      float bb = bias[n];
#pragma unroll
      for (int r = 0; r < 8; ++r) {
        int m = mBase + wm + mi * 16 + 8 * g + r;
        if (m >= rows) continue;
        float v = acc[mi][ni][r] + bb;
        if (EPI == 0) {
          Cf[(size_t)m * N + n] = v;
        } else if (EPI == 1) {
          Ch[(size_t)m * N + n] = (_Float16)(0.5f * v * (1.0f + erff(v * 0.70710678118654752f)));
        } else {
          Cf[(size_t)m * N + n] += v;
        }
      }
    }
}

// ---------------- LayerNorm (one wave per row), writes f16 ----------------
__global__ __launch_bounds__(256)
void k_ln(const float* __restrict__ X, const float* __restrict__ gw,
          const float* __restrict__ gb, _Float16* __restrict__ O, int rows) {
  int wid  = (blockIdx.x * blockDim.x + threadIdx.x) >> 5;
  int lane = threadIdx.x & 31;
  if (wid >= rows) return;
  const float* x = X + (size_t)wid * C_;
  float s = 0.f, s2 = 0.f;
  for (int c = lane; c < C_; c += 32) { float v = x[c]; s += v; s2 += v * v; }
  for (int o = 16; o > 0; o >>= 1) { s += __shfl_xor(s, o, 32); s2 += __shfl_xor(s2, o, 32); }
  float mu  = s * (1.0f / C_);
  float var = s2 * (1.0f / C_) - mu * mu;
  float rs  = rsqrtf(var + 1e-6f);
  _Float16* o = O + (size_t)wid * C_;
  for (int c = lane; c < C_; c += 32)
    o[c] = (_Float16)((x[c] - mu) * rs * gw[c] + gb[c]);
}

// ---------------- patch im2col (NCHW, 16x16 stride-16) -> f16 ----------------
__global__ void k_im2col(const float* __restrict__ x, _Float16* __restrict__ Ap) {
  int i = blockIdx.x * blockDim.x + threadIdx.x;
  if (i >= PROWS_ * C_) return;
  int row = i / C_, k = i % C_;
  int b = row / 196, p = row % 196;
  int py = p / 14, px = p % 14;
  int c = k >> 8, r = k & 255;
  int hy = r >> 4, hx = r & 15;
  Ap[i] = (_Float16)x[(((size_t)b * 3 + c) * 224 + py * 16 + hy) * 224 + px * 16 + hx];
}

// ---------------- assemble tokens: cls + pos, patches + pos ----------------
__global__ void k_assemble(const float* __restrict__ po, const float* __restrict__ cls,
                           const float* __restrict__ pos, float* __restrict__ t) {
  int i = blockIdx.x * blockDim.x + threadIdx.x;
  if (i >= ROWS_ * C_) return;
  int row = i / C_, c = i % C_;
  int b = row / NP_, j = row % NP_;
  float v = (j == 0) ? cls[c] : po[(size_t)(b * 196 + j - 1) * C_ + c];
  t[i] = v + pos[j * C_ + c];
}

// ---------------- performer features pass A: dd = data @ w_orf^T, diag, maxes
__global__ __launch_bounds__(256)
void k_featA(const float* __restrict__ qkv, const float* __restrict__ worf,
             float* __restrict__ ddq, float* __restrict__ ddk,
             float* __restrict__ diagq, float* __restrict__ diagk,
             float* __restrict__ qmax, unsigned* __restrict__ kmaxu) {
  int wid  = (blockIdx.x * blockDim.x + threadIdx.x) >> 5;
  int lane = threadIdx.x & 31;
  if (wid >= ROWS_ * H_) return;
  int row = wid / H_, h = wid % H_;
  const float* qd = qkv + (size_t)row * QKVN_ + h * DH_;
  const float* kd = qd + C_;
  const float* w0 = worf + lane * DH_;
  const float* w1 = worf + (lane + 32) * DH_;
  float aq0 = 0, aq1 = 0, ak0 = 0, ak1 = 0, sq = 0, sk = 0;
  for (int d = 0; d < DH_; ++d) {
    float qv = qd[d], kv = kd[d];
    sq += qv; sk += kv;
    float wa = w0[d], wb = w1[d];
    aq0 += qv * wa; aq1 += qv * wb;
    ak0 += kv * wa; ak1 += kv * wb;
  }
  float* dq = ddq + (size_t)wid * M_;
  float* dk = ddk + (size_t)wid * M_;
  dq[lane] = aq0; dq[lane + 32] = aq1;
  dk[lane] = ak0; dk[lane + 32] = ak1;
  float mq = fmaxf(aq0, aq1), mk = fmaxf(ak0, ak1);
  for (int o = 16; o > 0; o >>= 1) {
    mq = fmaxf(mq, __shfl_xor(mq, o, 32));
    mk = fmaxf(mk, __shfl_xor(mk, o, 32));
  }
  if (lane == 0) {
    diagq[wid] = sq * 0.0625f;   // sum/2 * dn^2, dn^2 = 1/8
    diagk[wid] = sk * 0.0625f;
    qmax[wid]  = mq;
    atomicMax(kmaxu, f2ord(mk)); // global stabilizer for k
  }
}

// ---------------- pass B: qp/kp = ratio*(exp(dd - diag - stab) + 1e-6) ------
__global__ void k_featB(float* __restrict__ ddq, float* __restrict__ ddk,
                        const float* __restrict__ diagq, const float* __restrict__ diagk,
                        const float* __restrict__ qmax, const unsigned* __restrict__ kmaxu) {
  int i = blockIdx.x * blockDim.x + threadIdx.x;
  if (i >= ROWS_ * H_ * M_) return;
  int wid = i >> 6;
  const float ratio = 0.35355339059327373f;   // 1 / M^0.25
  float stabk = ord2f(*kmaxu);
  ddq[i] = ratio * (expf(ddq[i] - diagq[wid] - qmax[wid]) + 1e-6f);
  ddk[i] = ratio * (expf(ddk[i] - diagk[wid] - stabk) + 1e-6f);
}

// ---------------- kv[b,h,m,d] = sum_l kp * v ; ks[b,h,m] = sum_l kp --------
__global__ __launch_bounds__(256)
void k_kv(const float* __restrict__ kp, const float* __restrict__ qkv,
          float* __restrict__ kv, float* __restrict__ ks) {
  int b = blockIdx.x / H_, h = blockIdx.x % H_;
  __shared__ float kps[16][64];
  __shared__ float vs[16][64];
  int tid = threadIdx.x;
  int m = tid >> 2, d0 = (tid & 3) * 16;
  float acc[16];
#pragma unroll
  for (int j = 0; j < 16; ++j) acc[j] = 0.f;
  float ksa = 0.f;
  for (int l0 = 0; l0 < NP_; l0 += 16) {
    for (int i = tid; i < 16 * 64; i += 256) {
      int ll = i >> 6, c = i & 63, l = l0 + ll;
      kps[ll][c] = (l < NP_) ? kp[(size_t)((b * NP_ + l) * H_ + h) * M_ + c] : 0.f;
      vs[ll][c]  = (l < NP_) ? qkv[(size_t)(b * NP_ + l) * QKVN_ + 2 * C_ + h * DH_ + c] : 0.f;
    }
    __syncthreads();
    for (int ll = 0; ll < 16; ++ll) {
      float a = kps[ll][m];
      if ((tid & 3) == 0) ksa += a;
#pragma unroll
      for (int j = 0; j < 16; ++j) acc[j] += a * vs[ll][d0 + j];
    }
    __syncthreads();
  }
  float* kvp = kv + (size_t)((b * H_ + h) * M_ + m) * DH_ + d0;
#pragma unroll
  for (int j = 0; j < 16; ++j) kvp[j] = acc[j];
  if ((tid & 3) == 0) ks[(size_t)(b * H_ + h) * M_ + m] = ksa;
}

// ---------------- out = (qp @ kv) / (qp . ks) -> f16, one wave per (row,h) --
__global__ __launch_bounds__(256)
void k_out(const float* __restrict__ qp, const float* __restrict__ kv,
           const float* __restrict__ ks, _Float16* __restrict__ out16) {
  int wid  = (blockIdx.x * blockDim.x + threadIdx.x) >> 5;
  int lane = threadIdx.x & 31;
  if (wid >= ROWS_ * H_) return;
  int row = wid / H_, h = wid % H_;
  int b = row / NP_;
  const float* q   = qp + (size_t)wid * M_;
  const float* kvb = kv + (size_t)(b * H_ + h) * M_ * DH_;
  const float* ksb = ks + (size_t)(b * H_ + h) * M_;
  float n0 = 0.f, n1 = 0.f, D = 0.f;
  for (int m = 0; m < M_; ++m) {
    float qm = q[m];
    D  += qm * ksb[m];
    n0 += qm * kvb[m * DH_ + lane];
    n1 += qm * kvb[m * DH_ + lane + 32];
  }
  _Float16* o = out16 + (size_t)row * C_ + h * DH_;
  o[lane]      = (_Float16)(n0 / D);
  o[lane + 32] = (_Float16)(n1 / D);
}

// ---------------- gather cls rows into compact f16 matrix -------------------
__global__ void k_gather_cls(const _Float16* __restrict__ ln16, _Float16* __restrict__ cls16) {
  int i = blockIdx.x * blockDim.x + threadIdx.x;
  if (i >= B_ * C_) return;
  int b = i / C_, c = i % C_;
  cls16[i] = ln16[(size_t)(b * NP_) * C_ + c];
}

// ============================================================================
extern "C" void kernel_launch(void* const* d_in, const int* in_sizes, int n_in,
                              void* d_out, int out_size, void* d_ws, size_t ws_size,
                              hipStream_t stream) {
  (void)in_sizes; (void)n_in; (void)out_size;
  const float* x       = (const float*)d_in[0];
  const float* patch_w = (const float*)d_in[1];
  const float* patch_b = (const float*)d_in[2];
  const float* cls_tok = (const float*)d_in[3];
  const float* pos     = (const float*)d_in[4];
  const float* ln1_w   = (const float*)d_in[5];
  const float* ln1_b   = (const float*)d_in[6];
  const float* qkv_w   = (const float*)d_in[7];
  const float* qkv_b   = (const float*)d_in[8];
  const float* w_orf   = (const float*)d_in[9];
  const float* proj_w  = (const float*)d_in[10];
  const float* proj_b  = (const float*)d_in[11];
  const float* ln2_w   = (const float*)d_in[12];
  const float* ln2_b   = (const float*)d_in[13];
  const float* fc1_w   = (const float*)d_in[14];
  const float* fc1_b   = (const float*)d_in[15];
  const float* fc2_w   = (const float*)d_in[16];
  const float* fc2_b   = (const float*)d_in[17];
  const float* lnf_w   = (const float*)d_in[18];
  const float* lnf_b   = (const float*)d_in[19];
  const float* head_w  = (const float*)d_in[20];
  const float* head_b  = (const float*)d_in[21];

  // ---- workspace layout (~138 MB total) ----
  char* ws = (char*)d_ws;
  size_t off = 0;
  auto take = [&](size_t bytes) { size_t r = off; off += (bytes + 255) & ~(size_t)255; return r; };
  float*    t_buf  = (float*)(ws + take((size_t)ROWS_ * C_ * 4));
  char*     qkvreg = ws + take((size_t)ROWS_ * QKVN_ * 4);    // 58 MB region
  float*    qkv    = (float*)qkvreg;
  float*    patcho = (float*)qkvreg;        // reused before attention exists
  _Float16* g16    = (_Float16*)qkvreg;     // reused after attention consumed
  char*     featrg = ws + take((size_t)ROWS_ * H_ * M_ * 4 * 2);
  float*    ddq    = (float*)featrg;
  float*    ddk    = (float*)(featrg + (size_t)ROWS_ * H_ * M_ * 4);
  _Float16* ap16   = (_Float16*)featrg;     // reused at patch-embed stage
  _Float16* act16  = (_Float16*)(ws + take((size_t)ROWS_ * C_ * 2));
  float*    kvb    = (float*)(ws + take((size_t)B_ * H_ * M_ * DH_ * 4));
  float*    ksb    = (float*)(ws + take((size_t)B_ * H_ * M_ * 4));
  float*    diagq  = (float*)(ws + take((size_t)ROWS_ * H_ * 4));
  float*    diagk  = (float*)(ws + take((size_t)ROWS_ * H_ * 4));
  float*    qmax   = (float*)(ws + take((size_t)ROWS_ * H_ * 4));
  unsigned* kmaxu  = (unsigned*)(ws + take(256));
  _Float16* w16    = (_Float16*)(ws + take((size_t)HID_ * C_ * 2)); // largest weight slab
  _Float16* cls16  = (_Float16*)(ws + take((size_t)B_ * C_ * 2));
  (void)ws_size;

  auto cvt = [&](const float* Wp, int n) {
    k_cvt_f16<<<(n + 255) / 256, 256, 0, stream>>>(Wp, w16, n);
  };

  // ---- patch embed: im2col + WMMA GEMM + assemble tokens ----
  cvt(patch_w, C_ * C_);
  k_im2col<<<(PROWS_ * C_ + 255) / 256, 256, 0, stream>>>(x, ap16);
  k_gemm<0><<<dim3(C_ / 128, PROWS_ / 128), 256, 0, stream>>>(
      ap16, w16, patch_b, patcho, nullptr, PROWS_, C_, C_);
  k_assemble<<<(ROWS_ * C_ + 255) / 256, 256, 0, stream>>>(patcho, cls_tok, pos, t_buf);

  const int lnGrid = (ROWS_ * 32 + 255) / 256;
  const int whGrid = (ROWS_ * H_ * 32 + 255) / 256;
  const int gy     = (ROWS_ + 127) / 128;   // 50 row-blocks

  for (int i = 0; i < L_; ++i) {
    // LN1 -> f16
    k_ln<<<lnGrid, 256, 0, stream>>>(t_buf, ln1_w + i * C_, ln1_b + i * C_, act16, ROWS_);
    // QKV GEMM
    cvt(qkv_w + (size_t)i * QKVN_ * C_, QKVN_ * C_);
    k_gemm<0><<<dim3(QKVN_ / 128, gy), 256, 0, stream>>>(
        act16, w16, qkv_b + i * QKVN_, qkv, nullptr, ROWS_, QKVN_, C_);
    // performer features
    hipMemsetAsync(kmaxu, 0, 4, stream);
    k_featA<<<whGrid, 256, 0, stream>>>(qkv, w_orf + i * M_ * DH_, ddq, ddk,
                                        diagq, diagk, qmax, kmaxu);
    k_featB<<<(ROWS_ * H_ * M_ + 255) / 256, 256, 0, stream>>>(ddq, ddk, diagq, diagk, qmax, kmaxu);
    k_kv<<<B_ * H_, 256, 0, stream>>>(ddk, qkv, kvb, ksb);
    k_out<<<whGrid, 256, 0, stream>>>(ddq, kvb, ksb, act16);
    // proj GEMM with residual accumulate into t
    cvt(proj_w + (size_t)i * C_ * C_, C_ * C_);
    k_gemm<2><<<dim3(C_ / 128, gy), 256, 0, stream>>>(
        act16, w16, proj_b + i * C_, t_buf, nullptr, ROWS_, C_, C_);
    // LN2 -> f16
    k_ln<<<lnGrid, 256, 0, stream>>>(t_buf, ln2_w + i * C_, ln2_b + i * C_, act16, ROWS_);
    // fc1 GEMM + exact GELU -> f16
    cvt(fc1_w + (size_t)i * HID_ * C_, HID_ * C_);
    k_gemm<1><<<dim3(HID_ / 128, gy), 256, 0, stream>>>(
        act16, w16, fc1_b + i * HID_, nullptr, g16, ROWS_, HID_, C_);
    // fc2 GEMM with residual accumulate into t
    cvt(fc2_w + (size_t)i * C_ * HID_, C_ * HID_);
    k_gemm<2><<<dim3(C_ / 128, gy), 256, 0, stream>>>(
        g16, w16, fc2_b + i * C_, t_buf, nullptr, ROWS_, C_, HID_);
  }

  // ---- final LN, gather cls rows, classifier head ----
  k_ln<<<lnGrid, 256, 0, stream>>>(t_buf, lnf_w, lnf_b, act16, ROWS_);
  k_gather_cls<<<(B_ * C_ + 255) / 256, 256, 0, stream>>>(act16, cls16);
  cvt(head_w, NCLS_ * C_);
  k_gemm<0><<<dim3((NCLS_ + 127) / 128, 1), 256, 0, stream>>>(
      cls16, w16, head_b, (float*)d_out, nullptr, B_, NCLS_, C_);
}
